// CADCConv2d_51058571215387
// MI455X (gfx1250) — matrix operand order repaired and checked by reading, not compiled, not run on hardware
//
#include <hip/hip_runtime.h>
#include <hip/hip_bf16.h>
#include <math.h>

// ---------------------------------------------------------------------------
// CADC deformable conv pipeline for MI455X (gfx1250), wave32 + WMMA bf16.
//
//   K1: 1x1 fuse conv   (M=64  x K=128 x N=131072)  -> x (f32, ws)
//       weights staged into LDS via the Tensor Data Mover (TDM).
//   K2: off/mod 3x3 conv(M=27  x K=576 x N=131072)  -> off, mod (f32, ws)
//   K3: bilinear-sample fused with regression GEMM
//       (M=64 x K=576 x N=131072), B built on the fly from gathers.
//
// All matrix math: v_wmma_f32_16x16x32_bf16 (bf16 in, f32 accumulate).
// ---------------------------------------------------------------------------

typedef __attribute__((ext_vector_type(16))) __bf16 v16bf;
typedef __attribute__((ext_vector_type(8)))  float  v8f;
typedef __attribute__((ext_vector_type(4)))  unsigned int u32x4;
typedef __attribute__((ext_vector_type(8)))  int  i32x8;
typedef __attribute__((ext_vector_type(4)))  int  i32x4;

#define NB 8
#define NC 64
#define HH 128
#define WW 128
#define HW (HH * WW)          // 16384
#define CHW (NC * HW)         // 64*16384
#define NPIX (NB * HW)        // 131072
#define KREG 576              // 64 channels * 9 taps

__device__ __forceinline__ v8f wmma_bf16(v16bf a, v16bf b, v8f c) {
  // (neg_a, A, neg_b, B, c_mod, C, reuse_a, reuse_b)
  return __builtin_amdgcn_wmma_f32_16x16x32_bf16(false, a, false, b,
                                                 (short)0, c, false, false);
}

// A-matrix fragment (16x32 bf16) from an LDS weight block, row-major [rows][ldk].
// k order used here (16-bit A 16x32, contiguous k; ISA table order differs): lane m=lane&15 holds row M=m;
// lanes 0-15: K = {0..15}; lanes 16-31: K = {16..31} (matches B fill).
__device__ __forceinline__ v16bf lds_a_frag(const __bf16* W, int M0, int K0,
                                            int ldk, int lane) {
  const int m = lane & 15, ksel = lane >> 4;
  v16bf a;
#pragma unroll
  for (int e = 0; e < 16; ++e) {
    const int k = K0 + ksel * 16 + e;
    a[e] = W[(M0 + m) * ldk + k];
  }
  return a;
}

__device__ __forceinline__ void zero_acc(v8f* acc, int n) {
  for (int i = 0; i < n; ++i)
#pragma unroll
    for (int e = 0; e < 8; ++e) acc[i][e] = 0.0f;
}

// ---------------------------------------------------------------------------
// TDM: DMA a contiguous run of nelem f32 (nelem <= 65535) from global into LDS.
// Builds a 2-D D# per ISA 08 (group0: count/lds_addr/global_addr/type=2;
// group1: data_size=4B, tensor_dim0 = tile_dim0 = nelem, tensor_dim1 =
// tile_dim1 = 1, stride0 = nelem).  Caller must s_wait_tensorcnt + barrier.
// This toolchain exposes the 6-arg builtin:
//   (u32x4 g0, i32x8 g1, i32x4 g2, i32x4 g3, i32x8 pad, i32 cpol)
// ---------------------------------------------------------------------------
__device__ __forceinline__ void tdm_load_1d_f32(const float* gsrc, void* lds_dst,
                                                unsigned nelem) {
  const unsigned lds_off = (unsigned)(size_t)lds_dst;        // low 32b = LDS byte offset
  const unsigned long long ga = (unsigned long long)(size_t)gsrc;
  u32x4 g0;
  g0[0] = 1u;                                   // count=1, user descriptor
  g0[1] = lds_off;                              // lds_addr
  g0[2] = (unsigned)ga;                         // global_addr[31:0]
  g0[3] = (unsigned)(ga >> 32) | (2u << 30);    // global_addr[56:32] | type=2
  i32x8 g1;
  g1[0] = (int)(2u << 16);                      // data_size = 4 bytes
  g1[1] = (int)((nelem & 0xFFFFu) << 16);       // tensor_dim0 lo16
  g1[2] = (int)((nelem >> 16) | (1u << 16));    // tensor_dim0 hi16 | tensor_dim1 = 1
  g1[3] = (int)((nelem & 0xFFFFu) << 16);       // tile_dim0
  g1[4] = 1;                                    // tile_dim1 = 1
  g1[5] = (int)nelem;                           // tensor_dim0_stride lo32
  g1[6] = 0;
  g1[7] = 0;
  i32x4 z4 = {0, 0, 0, 0};
  i32x8 z8 = {0, 0, 0, 0, 0, 0, 0, 0};
  __builtin_amdgcn_tensor_load_to_lds(g0, g1, z4, z4, z8, 0);
}

// ---------------------------------------------------------------------------
// Kernel 1: fuse 1x1 conv.  x[b,c,h,w] = sum_k W[c,k] * concat(img,cont)[b,k,h,w]
// ---------------------------------------------------------------------------
__global__ __launch_bounds__(256) void fuse_gemm_kernel(
    const float* __restrict__ x_img, const float* __restrict__ x_cont,
    const float* __restrict__ w_fuse, float* __restrict__ x_out) {
  __shared__ float  Wf32[64 * 128];    // TDM landing buffer (32 KB)
  __shared__ __bf16 Wlds[64 * 128];    // converted weights (16 KB)
  const int t = threadIdx.x;

  if (t < 32) {                        // wave 0 issues the tensor DMA
    tdm_load_1d_f32(w_fuse, Wf32, 64 * 128);
    __builtin_amdgcn_s_wait_tensorcnt(0);
  }
  __syncthreads();
  for (int i = t; i < 64 * 128; i += 256) Wlds[i] = (__bf16)Wf32[i];
  __syncthreads();

  const int lane = t & 31, wave = t >> 5;
  const long p0  = ((long)blockIdx.x * 8 + wave) * 16;   // 16-pixel tile
  const int  b   = (int)(p0 >> 14);
  const int  hw0 = (int)(p0 & (HW - 1));
  const float* imgb  = x_img  + (long)b * CHW + hw0;
  const float* contb = x_cont + (long)b * CHW + hw0;

  const int n = lane & 15, ksel = lane >> 4, msel = ksel;
  v8f acc[4];
  zero_acc(acc, 4);

#pragma unroll
  for (int kstep = 0; kstep < 4; ++kstep) {      // K = 128 in steps of 32
    const int K0 = kstep * 32;
    v16bf bfrag;
#pragma unroll
    for (int e = 0; e < 16; ++e) {               // B: K = K0 + 16*ksel + e
      const int k = K0 + ksel * 16 + e;
      const float v = (k < 64) ? imgb[(long)k * HW + n]
                               : contb[(long)(k - 64) * HW + n];
      bfrag[e] = (__bf16)v;
    }
#pragma unroll
    for (int mt = 0; mt < 4; ++mt) {
      v16bf afrag = lds_a_frag(Wlds, mt * 16, K0, 128, lane);
      acc[mt] = wmma_bf16(afrag, bfrag, acc[mt]);
    }
  }

  float* outp = x_out + (long)b * CHW + hw0 + n;
#pragma unroll
  for (int mt = 0; mt < 4; ++mt)
#pragma unroll
    for (int r = 0; r < 8; ++r) {                // C: M = 8*msel + r, N = n
      const int c = mt * 16 + msel * 8 + r;
      outp[(long)c * HW] = acc[mt][r];
    }
}

// ---------------------------------------------------------------------------
// Kernel 2: offsets (18ch) + modulation (9ch) 3x3 convs, one padded 32-row GEMM.
// ---------------------------------------------------------------------------
__global__ __launch_bounds__(256) void offmod_kernel(
    const float* __restrict__ x, const float* __restrict__ w_off,
    const float* __restrict__ b_off, const float* __restrict__ w_mod,
    const float* __restrict__ b_mod, float* __restrict__ off_out,
    float* __restrict__ mod_out) {
  __shared__ __bf16 Wlds[32 * KREG];             // rows 0-17 off, 18-26 mod, rest 0
  const int t = threadIdx.x;
  for (int i = t; i < 32 * KREG; i += 256) {
    const int co = i / KREG, k = i - co * KREG;
    float v = 0.0f;
    if (co < 18)      v = w_off[(long)co * KREG + k];
    else if (co < 27) v = w_mod[(long)(co - 18) * KREG + k];
    Wlds[i] = (__bf16)v;
  }
  __syncthreads();

  const int lane = t & 31, wave = t >> 5;
  const long p0  = ((long)blockIdx.x * 8 + wave) * 16;
  const int  b   = (int)(p0 >> 14);
  const int  hw0 = (int)(p0 & (HW - 1));
  const int  h   = hw0 >> 7, w0 = hw0 & (WW - 1);
  const float* xb = x + (long)b * CHW;

  const int n = lane & 15, ksel = lane >> 4, msel = ksel;
  v8f acc[2];
  zero_acc(acc, 2);

  for (int kstep = 0; kstep < 18; ++kstep) {     // K = 576 in steps of 32
    const int K0 = kstep * 32;
    v16bf bfrag;
#pragma unroll
    for (int e = 0; e < 16; ++e) {
      const int k   = K0 + ksel * 16 + e;        // k = ci*9 + tap
      const int ci  = k / 9, tap = k - ci * 9;
      const int ky  = tap / 3, kx = tap - ky * 3;
      const int y   = h + ky - 1;
      const int xx  = w0 + n + kx - 1;
      float v = 0.0f;
      if ((unsigned)y < (unsigned)HH && (unsigned)xx < (unsigned)WW)
        v = xb[(long)ci * HW + y * WW + xx];
      bfrag[e] = (__bf16)v;
    }
#pragma unroll
    for (int mt = 0; mt < 2; ++mt) {
      v16bf afrag = lds_a_frag(Wlds, mt * 16, K0, KREG, lane);
      acc[mt] = wmma_bf16(afrag, bfrag, acc[mt]);
    }
  }

#pragma unroll
  for (int mt = 0; mt < 2; ++mt)
#pragma unroll
    for (int r = 0; r < 8; ++r) {
      const int co = mt * 16 + msel * 8 + r;
      const float v = acc[mt][r];
      if (co < 18) {
        off_out[((long)b * 18 + co) * HW + hw0 + n] = v + b_off[co];
      } else if (co < 27) {
        const int cm = co - 18;
        const float s = v + b_mod[cm];
        mod_out[((long)b * 9 + cm) * HW + hw0 + n] = 2.0f / (1.0f + __expf(-s));
      }
    }
}

// ---------------------------------------------------------------------------
// Kernel 3: fused bilinear sampling + modulation + 64x576 regression GEMM.
// ---------------------------------------------------------------------------
struct alignas(16) SampleMeta {     // per (k2, pixel): 32 B
  float w00, w01, w10, w11;         // bilinear weights * mod, OOB folded to 0
  int   o00, o01, o10, o11;         // clamped flat offsets into an HxW plane
};

#define SMEM_W   (64 * KREG * 2)                 // 73728 B of bf16 w_reg
#define SMEM_M   (8 * 144 * (int)sizeof(SampleMeta))
#define SMEM_TOT (SMEM_W + SMEM_M)               // 110592 B dynamic LDS

__global__ __launch_bounds__(256) void deform_gemm_kernel(
    const float* __restrict__ x, const float* __restrict__ off_buf,
    const float* __restrict__ mod_buf, const float* __restrict__ w_reg,
    float* __restrict__ out) {
  extern __shared__ char smem_raw[];
  __bf16* Wlds = (__bf16*)smem_raw;                        // [64][576]
  SampleMeta* meta = (SampleMeta*)(smem_raw + SMEM_W);     // [8 waves][144]

  const int t = threadIdx.x;
  for (int i = t; i < 64 * KREG; i += 256) Wlds[i] = (__bf16)w_reg[i];

  const int lane = t & 31, wave = t >> 5;
  const long p0  = ((long)blockIdx.x * 8 + wave) * 16;
  const int  b   = (int)(p0 >> 14);
  const int  hw0 = (int)(p0 & (HW - 1));
  const int  h   = hw0 >> 7, w0 = hw0 & (WW - 1);

  // warm the gather-source neighborhood of this tile (global_prefetch_b8)
  __builtin_prefetch(x + (long)b * CHW + hw0, 0, 3);

  SampleMeta* my = meta + wave * 144;
  for (int idx = lane; idx < 144; idx += 32) {   // 9 taps x 16 pixels
    const int k2 = idx >> 4, n = idx & 15;
    const float dy = off_buf[((long)b * 18 + 2 * k2    ) * HW + hw0 + n];
    const float dx = off_buf[((long)b * 18 + 2 * k2 + 1) * HW + hw0 + n];
    const float m  = mod_buf[((long)b * 9 + k2) * HW + hw0 + n];
    const float py = dy + (float)(h - 1 + k2 / 3);
    const float px = dx + (float)(w0 + n - 1 + k2 % 3);
    const float fy = floorf(py), fx = floorf(px);
    const int y0 = (int)fy, x0 = (int)fx;
    const float wy1 = py - fy, wy0 = 1.0f - wy1;
    const float wx1 = px - fx, wx0 = 1.0f - wx1;
    const bool vy0 = (y0 >= 0) && (y0 < HH);
    const bool vy1 = (y0 + 1 >= 0) && (y0 + 1 < HH);
    const bool vx0 = (x0 >= 0) && (x0 < WW);
    const bool vx1 = (x0 + 1 >= 0) && (x0 + 1 < WW);
    const int y0c = y0     < 0 ? 0 : (y0     > HH - 1 ? HH - 1 : y0);
    const int y1c = y0 + 1 < 0 ? 0 : (y0 + 1 > HH - 1 ? HH - 1 : y0 + 1);
    const int x0c = x0     < 0 ? 0 : (x0     > WW - 1 ? WW - 1 : x0);
    const int x1c = x0 + 1 < 0 ? 0 : (x0 + 1 > WW - 1 ? WW - 1 : x0 + 1);
    SampleMeta sm;
    sm.w00 = wy0 * wx0 * m * (float)(vy0 && vx0);
    sm.w01 = wy0 * wx1 * m * (float)(vy0 && vx1);
    sm.w10 = wy1 * wx0 * m * (float)(vy1 && vx0);
    sm.w11 = wy1 * wx1 * m * (float)(vy1 && vx1);
    sm.o00 = y0c * WW + x0c;  sm.o01 = y0c * WW + x1c;
    sm.o10 = y1c * WW + x0c;  sm.o11 = y1c * WW + x1c;
    my[idx] = sm;
  }
  __syncthreads();

  const float* xb = x + (long)b * CHW;
  const int n = lane & 15, ksel = lane >> 4, msel = ksel;
  v8f acc[4];
  zero_acc(acc, 4);

  for (int kstep = 0; kstep < 18; ++kstep) {     // K = 576 in steps of 32
    const int K0 = kstep * 32;
    v16bf bfrag;
#pragma unroll
    for (int e = 0; e < 16; ++e) {
      const int k  = K0 + ksel * 16 + e;         // k = i*9 + k2
      const int i  = k / 9, k2 = k - i * 9;
      const SampleMeta sm = my[k2 * 16 + n];
      const float* ci = xb + (long)i * HW;
      const float v = sm.w00 * ci[sm.o00] + sm.w01 * ci[sm.o01]
                    + sm.w10 * ci[sm.o10] + sm.w11 * ci[sm.o11];
      bfrag[e] = (__bf16)v;
    }
#pragma unroll
    for (int mt = 0; mt < 4; ++mt) {
      v16bf afrag = lds_a_frag(Wlds, mt * 16, K0, KREG, lane);
      acc[mt] = wmma_bf16(afrag, bfrag, acc[mt]);
    }
  }

  float* outp = out + (long)b * CHW + hw0 + n;
#pragma unroll
  for (int mt = 0; mt < 4; ++mt)
#pragma unroll
    for (int r = 0; r < 8; ++r) {
      const int c = mt * 16 + msel * 8 + r;
      outp[(long)c * HW] = acc[mt][r];
    }
}

// ---------------------------------------------------------------------------
// Launcher.  Workspace layout (requires ws_size >= ~47.8 MB):
//   [0, 33554432)            x      f32 [8][64][128][128]
//   [33554432, 42991616)     off    f32 [8][18][128][128]
//   [42991616, 47710208)     mod    f32 [8][ 9][128][128]
// ---------------------------------------------------------------------------
extern "C" void kernel_launch(void* const* d_in, const int* in_sizes, int n_in,
                              void* d_out, int out_size, void* d_ws, size_t ws_size,
                              hipStream_t stream) {
  const float* x_img  = (const float*)d_in[0];
  const float* x_cont = (const float*)d_in[1];
  const float* w_fuse = (const float*)d_in[2];
  const float* w_off  = (const float*)d_in[3];
  const float* b_off  = (const float*)d_in[4];
  const float* w_mod  = (const float*)d_in[5];
  const float* b_mod  = (const float*)d_in[6];
  const float* w_reg  = (const float*)d_in[7];
  float* out = (float*)d_out;

  char* ws = (char*)d_ws;
  float* x_f32   = (float*)(ws);
  float* off_buf = (float*)(ws + (size_t)NB * CHW * sizeof(float));
  float* mod_buf = (float*)(ws + (size_t)NB * CHW * sizeof(float)
                               + (size_t)NB * 18 * HW * sizeof(float));

  const int blocks = NPIX / (16 * 8);            // 16 px/wave, 8 waves/block

  fuse_gemm_kernel<<<blocks, 256, 0, stream>>>(x_img, x_cont, w_fuse, x_f32);
  offmod_kernel<<<blocks, 256, 0, stream>>>(x_f32, w_off, b_off, w_mod, b_mod,
                                            off_buf, mod_buf);
  deform_gemm_kernel<<<blocks, 256, SMEM_TOT, stream>>>(x_f32, off_buf, mod_buf,
                                                        w_reg, out);
}
